// SurfaceVAE_FSQ_5901285065117
// MI455X (gfx1250) — matrix operand order repaired and checked by reading, not compiled, not run on hardware
//
#include <hip/hip_runtime.h>
#include <hip/hip_bf16.h>

typedef __attribute__((ext_vector_type(16))) _Float16 v16h;
typedef __attribute__((ext_vector_type(8)))  _Float16 v8h;
typedef __attribute__((ext_vector_type(8)))  float    v8f;
typedef __attribute__((ext_vector_type(4)))  int      v4i_t;

#define ROWS   64
#define AS     520      // activation LDS row stride in halves (bank-conflict padding)
#define NTHR   128

// ---- LDS layout, half region (offsets in halves)
#define H_X    0
#define H_Y    (ROWS*AS)
#define H_W    (H_Y + ROWS*AS)      // double-buffered weight slab: 2 * 8 tiles * 32 lanes * 16 halves
#define H_PEW  (H_W + 8192)         // 5*48*32 f16
#define H_DRW  (H_PEW + 7680)       // 5*32*48 f16
#define H_END  (H_DRW + 7680)
// float region (offsets in floats), base byte = 2*H_END
#define F_PEB  0
#define F_DRB  160
#define F_TEMB 400
#define F_PIW  480
#define F_PIB  992
#define F_POW  996
#define F_POB  1508
#define F_CLW  1636
#define F_CLB  2276
#define F_CDW  2281
#define F_CDB  2537
#define F_END  2544
#define SMEM_BYTES (2*H_END + 4*F_END + 4*ROWS)

// ---- workspace layout: f16 weights pre-packed in B-fragment slab order (offsets in halves)
// per layer: [chunk][ktile][tile(NCH)][lane(32)][half(16)]
#define WSO_E1 0                    // Kt=2,  N=512 -> 4*2*4096
#define WSO_E2 32768                // Kt=16, N=256 -> 2*16*4096
#define WSO_E3 163840               // Kt=8,  N=128 -> 8*4096
#define WSO_E4 196608               // Kt=4,  N=128 -> 4*4096
#define WSO_D1 212992               // Kt=5,  N=256 -> 2*5*4096
#define WSO_D2 253952               // Kt=8,  N=512 -> 4*8*4096
#define WSO_D3 385024               // Kt=16, N=32  -> 16*1024
#define WS_TOT 401408               // halves (802816 bytes)

// ---------------------------------------------------------------------------
// CDNA5 async global->LDS copy (ASYNCcnt). Builtin signature (probe-derived):
//   param0 = AS(1) int4*  (global src), param1 = AS(3) int4* (LDS dst),
//   param2 = imm offset, param3 = cpol.
// ---------------------------------------------------------------------------
__device__ __forceinline__ void async_cp16(const void* g, void* l) {
#if __has_builtin(__builtin_amdgcn_global_load_async_to_lds_b128)
  typedef __attribute__((address_space(1))) v4i_t* gv4p;
  typedef __attribute__((address_space(3))) v4i_t* lv4p;
  __builtin_amdgcn_global_load_async_to_lds_b128(
      (gv4p)(uintptr_t)g,           // inttoptr: global aperture is identity
      (lv4p)l,                      // addrspacecast generic->LDS
      0, 0);
#else
  unsigned lo = (unsigned)(uintptr_t)(__attribute__((address_space(3))) void*)l;
  unsigned long long ga = (unsigned long long)(uintptr_t)g;
  asm volatile("global_load_async_to_lds_b128 %0, %1, off"
               :: "v"(lo), "v"(ga) : "memory");
#endif
}

template <int N>
__device__ __forceinline__ void wait_async() {
#if __has_builtin(__builtin_amdgcn_s_wait_asynccnt)
  __builtin_amdgcn_s_wait_asynccnt(N);
#else
  asm volatile("s_wait_asynccnt %0" :: "n"(N) : "memory");
#endif
}

// ---------------------------------------------------------------------------
// Pre-pass: repack all layer weights f32 (K x N, row-major) -> f16 fragment
// slabs in d_ws. One thread per output half. Runs once per launch (~400k thr).
// ---------------------------------------------------------------------------
__global__ void __launch_bounds__(256) prep_weights_kernel(
    const float* __restrict__ e1w, const float* __restrict__ e2w,
    const float* __restrict__ e3w, const float* __restrict__ e4w,
    const float* __restrict__ d1w, const float* __restrict__ d2w,
    const float* __restrict__ d3w, _Float16* __restrict__ ws)
{
  const int i = blockIdx.x * 256 + threadIdx.x;
  if (i >= WS_TOT) return;
  const int off[8]   = {WSO_E1, WSO_E2, WSO_E3, WSO_E4, WSO_D1, WSO_D2, WSO_D3, WS_TOT};
  const int kreal[7] = {48, 512, 256, 128, 144, 256, 512};
  const int kt[7]    = {2, 16, 8, 4, 5, 8, 16};
  const int nn[7]    = {512, 256, 128, 128, 256, 512, 32};
  int l = 0;
  #pragma unroll
  for (int q = 0; q < 6; ++q) l += (i >= off[q + 1]) ? 1 : 0;
  const int li     = i - off[l];
  const int N      = nn[l];
  const int NCH    = (N >= 128) ? 8 : (N >> 4);
  const int roundH = NCH << 9;
  const int round  = li / roundH;
  const int j      = li % roundH;
  const int nt     = j >> 9;
  const int lane   = (j >> 4) & 31;
  const int h      = j & 15;
  const int chunk  = round / kt[l];
  const int kb     = round % kt[l];
  const int n = (chunk << 7) + (nt << 4) + (lane & 15);
  const int k = (kb << 5) + ((lane >> 4) << 4) + h;
  const float* W = (l == 0) ? e1w : (l == 1) ? e2w : (l == 2) ? e3w :
                   (l == 3) ? e4w : (l == 4) ? d1w : (l == 5) ? d2w : d3w;
  const float v = (k < kreal[l]) ? W[(size_t)k * N + n] : 0.f;
  ws[i] = (_Float16)v;
}

// ---------------------------------------------------------------------------
// One MLP layer via WMMA. Weights come from pre-packed ws slabs; staging is a
// double-buffered async global->LDS copy overlapped with compute.
// ---------------------------------------------------------------------------
__device__ __forceinline__ void gemm_layer(
    const _Float16* __restrict__ src, _Float16* __restrict__ dst,
    _Float16* __restrict__ sW,                   // 2 x 4096 halves in LDS
    const _Float16* __restrict__ wsL,            // layer base in packed ws
    const float* __restrict__ bias,
    int Ktiles, int Nout, bool relu, int tid)
{
  const int lane = tid & 31;
  const int wave = tid >> 5;
  const int m    = lane & 15;
  const int hi   = lane >> 4;
  const int rowbase = wave << 4;

  for (int nbase = 0; nbase < Nout; nbase += 128) {
    const int NCH    = ((Nout - nbase) >= 128) ? 8 : ((Nout - nbase) >> 4);
    const int roundH = NCH << 9;                 // halves per slab
    const int bytes  = roundH * 2;
    const _Float16* wchunk = wsL + (size_t)(nbase >> 7) * Ktiles * roundH;
    v8f acc[8] = {};

    // prefetch slab 0 -> buf0 (prior buf users are barrier-separated below)
    for (int o = tid * 16; o < bytes; o += NTHR * 16)
      async_cp16((const char*)wchunk + o, (char*)sW + o);

    for (int kb = 0; kb < Ktiles; ++kb) {
      const int cur = kb & 1;
      if (kb + 1 < Ktiles) {
        const char* gsrc = (const char*)(wchunk + (size_t)(kb + 1) * roundH);
        char*       ldst = (char*)(sW + (((kb + 1) & 1) << 12));
        for (int o = tid * 16; o < bytes; o += NTHR * 16)
          async_cp16(gsrc + o, ldst + o);
        // async loads complete in order: <=IPT outstanding => slab kb is done
        if (NCH == 8) wait_async<4>(); else wait_async<1>();
      } else {
        wait_async<0>();
      }
      __syncthreads();                            // slab kb visible block-wide

      const _Float16* sWc = sW + (cur << 12);
      // A fragment (16x32 f16, contiguous k = 16*hi + s): two adjacent 16B runs
      const _Float16* arow = src + (size_t)(rowbase + m) * AS + (kb << 5);
      v8h alo = *(const v8h*)(arow + hi * 16);
      v8h ahi = *(const v8h*)(arow + hi * 16 + 8);
      v16h a = __builtin_shufflevector(alo, ahi, 0,1,2,3,4,5,6,7,8,9,10,11,12,13,14,15);
      #pragma unroll
      for (int nt = 0; nt < 8; ++nt) {
        if (nt < NCH) {                           // block-uniform: EXEC stays all-ones
          v16h b = *(const v16h*)&sWc[((nt << 5) + lane) << 4];
          acc[nt] = __builtin_amdgcn_wmma_f32_16x16x32_f16(
              false, a, false, b, (short)0, acc[nt], false, false);
        }
      }
      __syncthreads();                            // all done with buf cur before overwrite
    }
    // epilogue: bias + activation, write f16 rows back to LDS
    for (int nt = 0; nt < NCH; ++nt) {
      const int col = nbase + (nt << 4) + m;
      const float bv = bias[col];
      #pragma unroll
      for (int r = 0; r < 8; ++r) {
        float v = acc[nt][r] + bv;
        if (relu) v = fmaxf(v, 0.f);
        dst[(size_t)(rowbase + r + (hi << 3)) * AS + col] = (_Float16)v;
      }
    }
  }
}

__global__ void __launch_bounds__(NTHR) surface_vae_fsq_kernel(
    const float* __restrict__ params, const int* __restrict__ stype_g,
    const float* __restrict__ temb_g,
    const float* __restrict__ pew_g, const float* __restrict__ peb_g,
    const float* __restrict__ e1b, const float* __restrict__ e2b,
    const float* __restrict__ e3b, const float* __restrict__ e4b,
    const float* __restrict__ piw, const float* __restrict__ pib,
    const float* __restrict__ pow_, const float* __restrict__ pob,
    const float* __restrict__ d1b, const float* __restrict__ d2b,
    const float* __restrict__ d3b,
    const float* __restrict__ clw, const float* __restrict__ clb,
    const float* __restrict__ cdw, const float* __restrict__ cdb,
    const float* __restrict__ drw_g, const float* __restrict__ drb_g,
    const _Float16* __restrict__ ws,
    float* __restrict__ out, int Btot)
{
  extern __shared__ __align__(16) char smem[];
  _Float16* sH   = (_Float16*)smem;
  _Float16* sX   = sH + H_X;
  _Float16* sY   = sH + H_Y;
  _Float16* sW   = sH + H_W;
  _Float16* sPEW = sH + H_PEW;
  _Float16* sDRW = sH + H_DRW;
  float* sF   = (float*)(smem + 2 * H_END);
  float* sPEB = sF + F_PEB;  float* sDRB = sF + F_DRB;  float* sTEMB = sF + F_TEMB;
  float* sPIW = sF + F_PIW;  float* sPIB = sF + F_PIB;
  float* sPOW = sF + F_POW;  float* sPOB = sF + F_POB;
  float* sCLW = sF + F_CLW;  float* sCLB = sF + F_CLB;
  float* sCDW = sF + F_CDW;  float* sCDB = sF + F_CDB;
  int*   sTYP = (int*)(sF + F_END);

  const int tid  = threadIdx.x;
  const int row0 = blockIdx.x * ROWS;

  const size_t Bz    = (size_t)Btot;
  const size_t o_msk = Bz * 48;
  const size_t o_cls = Bz * 96;
  const size_t o_cd  = o_cls + Bz * 5;
  const size_t o_zq  = o_cd  + Bz * 2;
  const size_t o_idx = o_zq  + Bz * 128;

  // ---- stage small / per-type weights into LDS
  for (int i = tid; i < 7680; i += NTHR) sPEW[i] = (_Float16)pew_g[i];
  for (int i = tid; i < 7680; i += NTHR) sDRW[i] = (_Float16)drw_g[i];
  for (int i = tid; i < 160;  i += NTHR) sPEB[i]  = peb_g[i];
  for (int i = tid; i < 240;  i += NTHR) sDRB[i]  = drb_g[i];
  for (int i = tid; i < 80;   i += NTHR) sTEMB[i] = temb_g[i];
  for (int i = tid; i < 512;  i += NTHR) sPIW[i]  = piw[i];
  if (tid < 4) sPIB[tid] = pib[tid];
  for (int i = tid; i < 512;  i += NTHR) sPOW[i]  = pow_[i];
  if (tid < 128) sPOB[tid] = pob[tid];
  for (int i = tid; i < 640;  i += NTHR) sCLW[i]  = clw[i];
  if (tid < 5) sCLB[tid] = clb[tid];
  for (int i = tid; i < 256;  i += NTHR) sCDW[i]  = cdw[i];
  if (tid < 2) sCDB[tid] = cdb[tid];
  for (int i = tid; i < ROWS; i += NTHR)
    sTYP[i] = (row0 + i < Btot) ? stype_g[row0 + i] : 0;
  __syncthreads();

  // ---- build x0 (64 rows x 64 cols): [param_emb(32) | type_emb(16) | zero-pad(16)]
  {
    const int row = tid >> 1;
    const int h   = tid & 1;
    const int gid = row0 + row;
    if (gid < Btot) {
      const int t = sTYP[row];
      float pv[48];
      const float* pr = params + (size_t)gid * 48;
      #pragma unroll
      for (int d = 0; d < 48; ++d) pv[d] = pr[d];
      for (int c = 0; c < 16; ++c) {
        const int col = (h << 4) + c;
        float acc = sPEB[t * 32 + col];
        for (int d = 0; d < 48; ++d)
          acc += pv[d] * (float)sPEW[(t * 48 + d) * 32 + col];
        sX[(size_t)row * AS + col] = (_Float16)acc;
      }
      if (h) {
        for (int e = 0; e < 16; ++e)
          sX[(size_t)row * AS + 32 + e] = (_Float16)sTEMB[t * 16 + e];
        for (int c = 48; c < 64; ++c) sX[(size_t)row * AS + c] = (_Float16)0.f;
      }
    }
  }
  __syncthreads();

  // ---- encoder (WMMA, async-staged weights)
  gemm_layer(sX, sY, sW, ws + WSO_E1, e1b,  2, 512, true,  tid); __syncthreads();
  gemm_layer(sY, sX, sW, ws + WSO_E2, e2b, 16, 256, true,  tid); __syncthreads();
  gemm_layer(sX, sY, sW, ws + WSO_E3, e3b,  8, 128, true,  tid); __syncthreads();
  gemm_layer(sY, sX, sW, ws + WSO_E4, e4b,  4, 128, false, tid); __syncthreads();

  // ---- FSQ bottleneck + heads (one thread per row); z in sX, dec input -> sY
  if (tid < ROWS) {
    const int row = tid;
    const int gid = row0 + row;
    if (gid < Btot) {
      const int t = sTYP[row];
      float zp[4] = {sPIB[0], sPIB[1], sPIB[2], sPIB[3]};
      for (int j = 0; j < 128; ++j) {
        const float zj = (float)sX[(size_t)row * AS + j];
        #pragma unroll
        for (int jj = 0; jj < 4; ++jj) zp[jj] += zj * sPIW[j * 4 + jj];
      }
      const float half_l[4] = {3.5035f, 2.002f, 2.002f, 2.002f};
      const float offs[4]   = {0.5f, 0.f, 0.f, 0.f};
      const float shft[4]   = {0.14369504f, 0.f, 0.f, 0.f}; // atanh(0.5/3.5035)
      const float half_w[4] = {4.f, 2.f, 2.f, 2.f};
      const float basis[4]  = {1.f, 8.f, 40.f, 200.f};
      float codes[4]; float fidx = 0.f;
      #pragma unroll
      for (int jj = 0; jj < 4; ++jj) {
        float b = tanhf(zp[jj] + shft[jj]) * half_l[jj] - offs[jj];
        float q = rintf(b);                     // round-half-even == jnp.round
        codes[jj] = q / half_w[jj];
        fidx += (q + half_w[jj]) * basis[jj];
      }
      out[o_idx + gid] = (float)(int)fidx;
      float cl[5] = {sCLB[0], sCLB[1], sCLB[2], sCLB[3], sCLB[4]};
      float cd[2] = {sCDB[0], sCDB[1]};
      for (int j = 0; j < 128; ++j) {
        float s = sPOB[j];
        #pragma unroll
        for (int jj = 0; jj < 4; ++jj) s += codes[jj] * sPOW[jj * 128 + j];
        out[o_zq + (size_t)gid * 128 + j] = s;
        sY[(size_t)row * AS + j] = (_Float16)s;
        #pragma unroll
        for (int c = 0; c < 5; ++c) cl[c] += s * sCLW[j * 5 + c];
        cd[0] += s * sCDW[j * 2 + 0];
        cd[1] += s * sCDW[j * 2 + 1];
      }
      #pragma unroll
      for (int c = 0; c < 5; ++c) out[o_cls + (size_t)gid * 5 + c] = cl[c];
      out[o_cd + (size_t)gid * 2 + 0] = cd[0];
      out[o_cd + (size_t)gid * 2 + 1] = cd[1];
      for (int e = 0; e < 16; ++e)
        sY[(size_t)row * AS + 128 + e] = (_Float16)sTEMB[t * 16 + e];
      for (int c = 144; c < 160; ++c) sY[(size_t)row * AS + c] = (_Float16)0.f;
    }
  }
  __syncthreads();

  // ---- decoder (WMMA)
  gemm_layer(sY, sX, sW, ws + WSO_D1, d1b,  5, 256, true,  tid); __syncthreads();
  gemm_layer(sX, sY, sW, ws + WSO_D2, d2b,  8, 512, true,  tid); __syncthreads();
  gemm_layer(sY, sX, sW, ws + WSO_D3, d3b, 16,  32, false, tid); __syncthreads();

  // ---- per-type output projection + mask; pdec in sX cols 0..31
  if (tid < ROWS) {
    const int row = tid;
    const int gid = row0 + row;
    if (gid < Btot) {
      const int t = sTYP[row];
      float pd[32];
      #pragma unroll
      for (int p = 0; p < 32; ++p) pd[p] = (float)sX[(size_t)row * AS + p];
      const int rd = (t == 0) ? 8 : (t == 1) ? 16 : (t == 2) ? 24 : (t == 3) ? 32 : 48;
      for (int d = 0; d < 48; ++d) {
        float a = sDRB[t * 48 + d];
        for (int p = 0; p < 32; ++p)
          a += pd[p] * (float)sDRW[(t * 32 + p) * 48 + d];
        out[(size_t)gid * 48 + d]         = a;
        out[o_msk + (size_t)gid * 48 + d] = (d < rd) ? 1.f : 0.f;
      }
    }
  }
}

extern "C" void kernel_launch(void* const* d_in, const int* in_sizes, int n_in,
                              void* d_out, int out_size, void* d_ws, size_t ws_size,
                              hipStream_t stream) {
  (void)n_in; (void)out_size; (void)ws_size;   // ws needs >= 802816 bytes
  const int Btot = in_sizes[1];                // surface_type count == B
  _Float16* ws = (_Float16*)d_ws;

  // pre-pass: pack weights into fragment-ordered f16 slabs in workspace
  prep_weights_kernel<<<(WS_TOT + 255) / 256, 256, 0, stream>>>(
      (const float*)d_in[5],  (const float*)d_in[7],  (const float*)d_in[9],
      (const float*)d_in[11], (const float*)d_in[17], (const float*)d_in[19],
      (const float*)d_in[21], ws);

  (void)hipFuncSetAttribute(reinterpret_cast<const void*>(surface_vae_fsq_kernel),
                            hipFuncAttributeMaxDynamicSharedMemorySize, (int)SMEM_BYTES);
  const int grid = (Btot + ROWS - 1) / ROWS;
  surface_vae_fsq_kernel<<<grid, NTHR, SMEM_BYTES, stream>>>(
      (const float*)d_in[0],  (const int*)d_in[1],   (const float*)d_in[2],
      (const float*)d_in[3],  (const float*)d_in[4],
      (const float*)d_in[6],  (const float*)d_in[8],
      (const float*)d_in[10], (const float*)d_in[12],
      (const float*)d_in[13], (const float*)d_in[14],
      (const float*)d_in[15], (const float*)d_in[16],
      (const float*)d_in[18], (const float*)d_in[20],
      (const float*)d_in[22],
      (const float*)d_in[23], (const float*)d_in[24],
      (const float*)d_in[25], (const float*)d_in[26],
      (const float*)d_in[27], (const float*)d_in[28],
      ws, (float*)d_out, Btot);
}
